// BertBlock_10153302687795
// MI455X (gfx1250) — matrix operand order repaired and checked by reading, compiled, not run on hardware
//
#include <hip/hip_runtime.h>
#include <hip/hip_bf16.h>

// ---------------------------------------------------------------------------
// BERT block forward for gfx1250 (MI455X).  bf16 WMMA (V_WMMA_F32_16X16X32_BF16)
// for every GEMM + flash attention; f32 accumulation everywhere.
// GEMM main loop is double-buffered (register-staged) with global_prefetch
// hints so VMEM overlaps the WMMA pipeline.
// ---------------------------------------------------------------------------

typedef __attribute__((ext_vector_type(2)))  __bf16 v2bf;
typedef __attribute__((ext_vector_type(8)))  __bf16 v8bf;
typedef __attribute__((ext_vector_type(16))) __bf16 v16bf;
typedef __attribute__((ext_vector_type(8)))  float  v8f;

#define N_EMBD   768
#define N_HEAD   12
#define HEAD_DIM 64
#define BATCH    4
#define SEQ      2048
#define M_ROWS   (BATCH * SEQ)          // 8192

__device__ __forceinline__ v8f v8f_zero() {
  v8f z = {0.f, 0.f, 0.f, 0.f, 0.f, 0.f, 0.f, 0.f};
  return z;
}

__device__ __forceinline__ v16bf cat8(v8bf lo, v8bf hi) {
  return __builtin_shufflevector(lo, hi, 0, 1, 2, 3, 4, 5, 6, 7,
                                         8, 9, 10, 11, 12, 13, 14, 15);
}

// A fragment (16 x 32 bf16, row-major source, rowStride in halves).
// ISA layout: lane l -> M = l&15 ; element j -> K = 16*(j>>3) + 8*(l>>4) + (j&7)
__device__ __forceinline__ v16bf load_fragA(const __bf16* base, int lane, int rowStride) {
  const int r  = lane & 15;
  const int kb = (lane >> 4) * 8;
  const __bf16* p = base + r * rowStride + kb;
  return cat8(*(const v8bf*)(p), *(const v8bf*)(p + 16));
}

// B fragment (32 x 16 bf16) from a TRANSPOSED tile baseT[n][k] (rowStride halves).
// ISA layout: lane l -> N = l&15 ; element j -> K = 16*(j>>3) + 8*(l>>4) + (j&7)
__device__ __forceinline__ v16bf load_fragB(const __bf16* baseT, int lane, int rowStride) {
  const int n  = lane & 15;
  const int kb = (lane >> 4) * 8;
  const __bf16* p = baseT + n * rowStride + kb;
  return cat8(*(const v8bf*)(p), *(const v8bf*)(p + 16));
}

__device__ __forceinline__ v8f wmma_bf16(v16bf a, v16bf b, v8f c) {
  return __builtin_amdgcn_wmma_f32_16x16x32_bf16(false, a, false, b,
                                                 (short)0, c, false, false);
}

__device__ __forceinline__ float gelu_tanh(float x) {
  const float c = 0.7978845608028654f;  // sqrt(2/pi)
  float u = c * (x + 0.044715f * x * x * x);
  return 0.5f * x * (1.0f + tanhf(u));
}

// ---------------------------------------------------------------------------
// fp32 -> bf16 conversion (weights)
// ---------------------------------------------------------------------------
__global__ void f32_to_bf16_kernel(const float* __restrict__ in,
                                   __bf16* __restrict__ out, int n) {
  int i = blockIdx.x * blockDim.x + threadIdx.x;
  if (i < n) out[i] = (__bf16)in[i];
}

// ---------------------------------------------------------------------------
// LayerNorm: f32 [rows, C] -> bf16 [rows, C]
// ---------------------------------------------------------------------------
__global__ __launch_bounds__(256)
void layernorm_bf16_kernel(const float* __restrict__ x,
                           const float* __restrict__ g,
                           const float* __restrict__ b,
                           __bf16* __restrict__ out, int C) {
  const int row = blockIdx.x;
  const float* xr = x + (size_t)row * C;
  float s = 0.f, s2 = 0.f;
  for (int i = threadIdx.x; i < C; i += blockDim.x) {
    float v = xr[i];
    s += v;
    s2 += v * v;
  }
  __shared__ float red0[256];
  __shared__ float red1[256];
  red0[threadIdx.x] = s;
  red1[threadIdx.x] = s2;
  __syncthreads();
  for (int off = 128; off > 0; off >>= 1) {
    if ((int)threadIdx.x < off) {
      red0[threadIdx.x] += red0[threadIdx.x + off];
      red1[threadIdx.x] += red1[threadIdx.x + off];
    }
    __syncthreads();
  }
  const float mean = red0[0] / (float)C;
  const float var  = red1[0] / (float)C - mean * mean;
  const float inv  = rsqrtf(var + 1e-5f);
  __bf16* orow = out + (size_t)row * C;
  for (int i = threadIdx.x; i < C; i += blockDim.x)
    orow[i] = (__bf16)((xr[i] - mean) * inv * g[i] + b[i]);
}

// ---------------------------------------------------------------------------
// Tiled bf16 WMMA GEMM:  out[M,N] = A[M,K] @ B[K,N] + bias  (+epilogue)
// Block: 256 threads (8 waves), tile 128x128, BK=32, double-buffered LDS.
// Wave grid 2(M) x 4(N): each wave computes 64x32 via 4x2 16x16 accumulators.
// Requires M%128==0, N%128==0, K%32==0.
// ---------------------------------------------------------------------------
enum { EPI_BIAS_BF16 = 0, EPI_BIAS_RES_F32 = 1, EPI_BIAS_GELU_BF16 = 2 };

#define LDS_STRIDE 40  // 32 + 8 pad halves; 80B rows keep 16B alignment

template <int EPI>
__global__ __launch_bounds__(256)
void gemm_bf16_kernel(const __bf16* __restrict__ A,
                      const __bf16* __restrict__ Bm,
                      const float* __restrict__ bias,
                      const float* __restrict__ residual,
                      float* __restrict__ outF,
                      __bf16* __restrict__ outB,
                      int M, int N, int K) {
  const int t = threadIdx.x;
  const int lane = t & 31;
  const int wave = t >> 5;
  const int wm = wave & 1;   // 0..1
  const int wn = wave >> 1;  // 0..3
  const int m0 = blockIdx.y * 128;
  const int n0 = blockIdx.x * 128;

  __shared__ __bf16 As[2][128 * LDS_STRIDE];
  __shared__ __bf16 BsT[2][128 * LDS_STRIDE];  // transposed: [n][k]

  // --- per-thread staging assignment ---
  // A tile (128 rows x 32 k): thread -> row t>>1, 16-half segment t&1
  const int arow = t >> 1, aseg = t & 1;
  // B tile (32 k x 128 n): thread owns k-pair (2*bkp, 2*bkp+1) x 8 n-values,
  // so transposed LDS stores merge into b32 writes.
  const int bkp = t & 15;   // k-pair index
  const int bns = t >> 4;   // n-segment (8 cols each)

  const __bf16* aP  = A  + (size_t)(m0 + arow) * K + aseg * 16;
  const __bf16* bP0 = Bm + (size_t)(2 * bkp) * N + n0 + bns * 8;

  v8bf aR0, aR1, bR0, bR1;
  auto fetch = [&](int kb) {
    const __bf16* pa = aP + kb;
    aR0 = *(const v8bf*)(pa);
    aR1 = *(const v8bf*)(pa + 8);
    const __bf16* pb = bP0 + (size_t)kb * N;
    bR0 = *(const v8bf*)(pb);        // row kb + 2*bkp
    bR1 = *(const v8bf*)(pb + N);    // row kb + 2*bkp + 1
  };
  auto store_lds = [&](int buf) {
    __bf16* da = &As[buf][arow * LDS_STRIDE + aseg * 16];
    *(v8bf*)(da) = aR0;
    *(v8bf*)(da + 8) = aR1;
#pragma unroll
    for (int i = 0; i < 8; ++i) {
      v2bf p;
      p[0] = bR0[i];
      p[1] = bR1[i];
      *(v2bf*)(&BsT[buf][(bns * 8 + i) * LDS_STRIDE + 2 * bkp]) = p;
    }
  };

  v8f acc[4][2];
#pragma unroll
  for (int i = 0; i < 4; ++i)
#pragma unroll
    for (int j = 0; j < 2; ++j) acc[i][j] = v8f_zero();

  fetch(0);
  store_lds(0);
  __syncthreads();

  int cur = 0;
  for (int kb = 0; kb < K; kb += 32) {
    const bool hasNext = (kb + 32) < K;
    if (hasNext) fetch(kb + 32);                       // overlap VMEM with WMMA
    if (kb + 128 < K) {                                // far prefetch -> L2
      __builtin_prefetch(aP + kb + 128, 0, 1);
      __builtin_prefetch(bP0 + (size_t)(kb + 128) * N, 0, 1);
    }

    v16bf af[4], bfr[2];
#pragma unroll
    for (int i = 0; i < 4; ++i)
      af[i] = load_fragA(&As[cur][(wm * 64 + i * 16) * LDS_STRIDE], lane, LDS_STRIDE);
#pragma unroll
    for (int j = 0; j < 2; ++j)
      bfr[j] = load_fragB(&BsT[cur][(wn * 32 + j * 16) * LDS_STRIDE], lane, LDS_STRIDE);
#pragma unroll
    for (int i = 0; i < 4; ++i)
#pragma unroll
      for (int j = 0; j < 2; ++j)
        acc[i][j] = wmma_bf16(af[i], bfr[j], acc[i][j]);

    if (hasNext) {
      store_lds(cur ^ 1);   // other buffer: no reader this iteration
      __syncthreads();
      cur ^= 1;
    }
  }

  // --- epilogue.  C layout: VGPR r -> row = r + 8*(lane>>4), col = lane&15 ---
  const int halfsel = lane >> 4;
  const int col_l = lane & 15;
#pragma unroll
  for (int j = 0; j < 2; ++j) {
    const int col = n0 + wn * 32 + j * 16 + col_l;
    const float bv = bias[col];
#pragma unroll
    for (int i = 0; i < 4; ++i) {
#pragma unroll
      for (int r = 0; r < 8; ++r) {
        const int row = m0 + wm * 64 + i * 16 + r + 8 * halfsel;
        const size_t idx = (size_t)row * N + col;
        float v = acc[i][j][r] + bv;
        if (EPI == EPI_BIAS_RES_F32) {
          outF[idx] = v + residual[idx];
        } else if (EPI == EPI_BIAS_GELU_BF16) {
          outB[idx] = (__bf16)gelu_tanh(v);
        } else {
          outB[idx] = (__bf16)v;
        }
      }
    }
  }
}

// ---------------------------------------------------------------------------
// Flash attention.  qkv: bf16 [B*T, 3C]; q/k/v column offsets 0/C/2C + h*64.
// Grid: (T/64, B*H); block 128 threads = 4 waves; wave owns 16 query rows.
// Scores via WMMA (q A-frags from global, k rows ARE the B-frag layout),
// online softmax in f32, P@V via WMMA with V transposed in LDS.
// ---------------------------------------------------------------------------
__global__ __launch_bounds__(128)
void attention_kernel(const __bf16* __restrict__ qkv,
                      const float* __restrict__ mask,
                      __bf16* __restrict__ y) {
  const int C = N_EMBD, T = SEQ, H = N_HEAD, D = HEAD_DIM;
  const int stride = 3 * C;  // 2304 halves per qkv row
  const int bh = blockIdx.y;
  const int b = bh / H, h = bh % H;
  const int t = threadIdx.x;
  const int lane = t & 31;
  const int wave = t >> 5;
  const int halfsel = lane >> 4;
  const int col_l = lane & 15;
  const int qrow0 = blockIdx.x * 64 + wave * 16;

  const __bf16* qkvB = qkv + (size_t)b * T * stride;

  __shared__ __bf16 Vt[64 * 72];        // transposed V tile: [d][key], pad 8
  __shared__ __bf16 Pbuf[4][16 * 72];   // per-wave P staging (C-layout -> A-layout)

  // q fragments for this wave's 16 rows (K split: d 0..31, 32..63)
  const __bf16* qbase = qkvB + (size_t)qrow0 * stride + h * D;
  const v16bf qa0 = load_fragA(qbase + 0, lane, stride);
  const v16bf qa1 = load_fragA(qbase + 32, lane, stride);

  v8f O[4];
#pragma unroll
  for (int j = 0; j < 4; ++j) O[j] = v8f_zero();
  float mrow[8], lrow[8];
#pragma unroll
  for (int r = 0; r < 8; ++r) {
    mrow[r] = -1e30f;
    lrow[r] = 0.f;
  }
  const float sm_scale = 0.125f;  // 1/sqrt(64)

  for (int kb0 = 0; kb0 < T; kb0 += 64) {
    __syncthreads();  // previous iteration done with Vt
    // cooperative transposed V-tile load (64 keys x 64 d), key-pairs -> b32 stores
    {
      const int kp = t & 31;           // key pair (2*kp, 2*kp+1)
      const int dseg = (t >> 5) * 16;  // 16 d-values
      const __bf16* v0p =
          qkvB + (size_t)(kb0 + 2 * kp) * stride + 2 * C + h * D + dseg;
      v8bf r0a = *(const v8bf*)(v0p);
      v8bf r0b = *(const v8bf*)(v0p + 8);
      v8bf r1a = *(const v8bf*)(v0p + stride);
      v8bf r1b = *(const v8bf*)(v0p + stride + 8);
#pragma unroll
      for (int i = 0; i < 8; ++i) {
        v2bf p;
        p[0] = r0a[i];
        p[1] = r1a[i];
        *(v2bf*)(&Vt[(dseg + i) * 72 + 2 * kp]) = p;
      }
#pragma unroll
      for (int i = 0; i < 8; ++i) {
        v2bf p;
        p[0] = r0b[i];
        p[1] = r1b[i];
        *(v2bf*)(&Vt[(dseg + 8 + i) * 72 + 2 * kp]) = p;
      }
    }
    __syncthreads();

    // --- scores S = scale * q @ k^T + mask_add  (16 q x 64 keys) ---
    v8f S[4];
#pragma unroll
    for (int ks = 0; ks < 4; ++ks) {
      const __bf16* kbase = qkvB + (size_t)(kb0 + ks * 16) * stride + C + h * D;
      v16bf kb_0 = load_fragB(kbase + 0, lane, stride);   // keys are B rows already
      v16bf kb_1 = load_fragB(kbase + 32, lane, stride);
      v8f s = v8f_zero();
      s = wmma_bf16(qa0, kb_0, s);
      s = wmma_bf16(qa1, kb_1, s);
      const float madd =
          (1.0f - mask[(size_t)b * T + kb0 + ks * 16 + col_l]) * -10000.0f;
#pragma unroll
      for (int r = 0; r < 8; ++r) S[ks][r] = s[r] * sm_scale + madd;
    }

    // --- online softmax update ---
#pragma unroll
    for (int r = 0; r < 8; ++r) {
      float m = fmaxf(fmaxf(S[0][r], S[1][r]), fmaxf(S[2][r], S[3][r]));
      m = fmaxf(m, __shfl_xor(m, 1));
      m = fmaxf(m, __shfl_xor(m, 2));
      m = fmaxf(m, __shfl_xor(m, 4));
      m = fmaxf(m, __shfl_xor(m, 8));
      const float mnew = fmaxf(mrow[r], m);
      const float corr = expf(mrow[r] - mnew);
      mrow[r] = mnew;
      float sum = 0.f;
#pragma unroll
      for (int ks = 0; ks < 4; ++ks) {
        float p = expf(S[ks][r] - mnew);
        S[ks][r] = p;
        sum += p;
      }
      sum += __shfl_xor(sum, 1);
      sum += __shfl_xor(sum, 2);
      sum += __shfl_xor(sum, 4);
      sum += __shfl_xor(sum, 8);
      lrow[r] = lrow[r] * corr + sum;
#pragma unroll
      for (int j = 0; j < 4; ++j) O[j][r] *= corr;
    }

    // --- P (C-layout regs) -> bf16 LDS (row-major) ---
    __bf16* Pw = Pbuf[wave];
#pragma unroll
    for (int ks = 0; ks < 4; ++ks)
#pragma unroll
      for (int r = 0; r < 8; ++r)
        Pw[(r + 8 * halfsel) * 72 + ks * 16 + col_l] = (__bf16)S[ks][r];
    __syncthreads();

    // --- O += P @ V_tile ---
    const v16bf pa0 = load_fragA(Pw + 0, lane, 72);   // keys 0..31
    const v16bf pa1 = load_fragA(Pw + 32, lane, 72);  // keys 32..63
#pragma unroll
    for (int j = 0; j < 4; ++j) {
      v16bf vb0 = load_fragB(Vt + (j * 16) * 72 + 0, lane, 72);
      v16bf vb1 = load_fragB(Vt + (j * 16) * 72 + 32, lane, 72);
      O[j] = wmma_bf16(pa0, vb0, O[j]);
      O[j] = wmma_bf16(pa1, vb1, O[j]);
    }
  }

  // --- normalize and write y in [B*T, C] (bf16 for the proj GEMM) ---
#pragma unroll
  for (int j = 0; j < 4; ++j)
#pragma unroll
    for (int r = 0; r < 8; ++r) {
      const int row = qrow0 + r + 8 * halfsel;
      const float ov = O[j][r] / lrow[r];
      y[((size_t)b * T + row) * C + h * D + j * 16 + col_l] = (__bf16)ov;
    }
}

// ---------------------------------------------------------------------------
// Host-side launch
// ---------------------------------------------------------------------------
extern "C" void kernel_launch(void* const* d_in, const int* in_sizes, int n_in,
                              void* d_out, int out_size, void* d_ws, size_t ws_size,
                              hipStream_t stream) {
  (void)in_sizes; (void)n_in; (void)out_size; (void)ws_size;
  const int C = N_EMBD, M = M_ROWS;

  const float* x         = (const float*)d_in[0];
  const float* attn_mask = (const float*)d_in[1];
  const float* ln1_g     = (const float*)d_in[2];
  const float* ln1_b     = (const float*)d_in[3];
  const float* ln2_g     = (const float*)d_in[4];
  const float* ln2_b     = (const float*)d_in[5];
  const float* W_attn    = (const float*)d_in[6];
  const float* b_attn    = (const float*)d_in[7];
  const float* W_proj    = (const float*)d_in[8];
  const float* b_proj    = (const float*)d_in[9];
  const float* W_fc      = (const float*)d_in[10];
  const float* b_fc      = (const float*)d_in[11];
  const float* W_out     = (const float*)d_in[12];
  const float* b_out     = (const float*)d_in[13];
  float* out             = (float*)d_out;

  // ---- workspace layout (bytes) ----
  char* ws = (char*)d_ws;
  size_t o = 0;
  __bf16* WAb = (__bf16*)(ws + o); o += (size_t)C * 3 * C * 2;        // 3.5 MB
  __bf16* WPb = (__bf16*)(ws + o); o += (size_t)C * C * 2;            // 1.2 MB
  __bf16* WFb = (__bf16*)(ws + o); o += (size_t)C * 4 * C * 2;        // 4.7 MB
  __bf16* WOb = (__bf16*)(ws + o); o += (size_t)4 * C * C * 2;        // 4.7 MB
  __bf16* Hb  = (__bf16*)(ws + o); o += (size_t)M * C * 2;            // 12.6 MB (ln out, reused)
  __bf16* QKVb = (__bf16*)(ws + o); o += (size_t)M * 4 * C * 2;       // 50.3 MB (qkv, reused for fc act)
  __bf16* Yb  = (__bf16*)(ws + o); o += (size_t)M * C * 2;            // 12.6 MB (attention out)
  float*  X2  = (float*)(ws + o);  o += (size_t)M * C * 4;            // 25.2 MB (post-attn residual)
  __bf16* FCb = QKVb;  // qkv dead after attention

  // ---- 1. weights fp32 -> bf16 ----
  {
    const int n1 = C * 3 * C, n2 = C * C, n3 = C * 4 * C, n4 = 4 * C * C;
    f32_to_bf16_kernel<<<(n1 + 255) / 256, 256, 0, stream>>>(W_attn, WAb, n1);
    f32_to_bf16_kernel<<<(n2 + 255) / 256, 256, 0, stream>>>(W_proj, WPb, n2);
    f32_to_bf16_kernel<<<(n3 + 255) / 256, 256, 0, stream>>>(W_fc,   WFb, n3);
    f32_to_bf16_kernel<<<(n4 + 255) / 256, 256, 0, stream>>>(W_out,  WOb, n4);
  }

  // ---- 2. ln1: x -> Hb (bf16) ----
  layernorm_bf16_kernel<<<M, 256, 0, stream>>>(x, ln1_g, ln1_b, Hb, C);

  // ---- 3. qkv = Hb @ W_attn + b_attn  -> bf16 [M, 3C] ----
  gemm_bf16_kernel<EPI_BIAS_BF16>
      <<<dim3(3 * C / 128, M / 128), 256, 0, stream>>>(
          Hb, WAb, b_attn, nullptr, nullptr, QKVb, M, 3 * C, C);

  // ---- 4. flash attention -> Yb (bf16 [M, C]) ----
  attention_kernel<<<dim3(SEQ / 64, BATCH * N_HEAD), 128, 0, stream>>>(
      QKVb, attn_mask, Yb);

  // ---- 5. x2 = x + Yb @ W_proj + b_proj  -> f32 ----
  gemm_bf16_kernel<EPI_BIAS_RES_F32>
      <<<dim3(C / 128, M / 128), 256, 0, stream>>>(
          Yb, WPb, b_proj, x, X2, nullptr, M, C, C);

  // ---- 6. ln2: X2 -> Hb (bf16) ----
  layernorm_bf16_kernel<<<M, 256, 0, stream>>>(X2, ln2_g, ln2_b, Hb, C);

  // ---- 7. fc = gelu(Hb @ W_fc + b_fc) -> bf16 [M, 4C] ----
  gemm_bf16_kernel<EPI_BIAS_GELU_BF16>
      <<<dim3(4 * C / 128, M / 128), 256, 0, stream>>>(
          Hb, WFb, b_fc, nullptr, nullptr, FCb, M, 4 * C, C);

  // ---- 8. out = X2 + fc @ W_out + b_out -> f32 (d_out) ----
  gemm_bf16_kernel<EPI_BIAS_RES_F32>
      <<<dim3(C / 128, M / 128), 256, 0, stream>>>(
          FCb, WOb, b_out, X2, out, nullptr, M, C, 4 * C);
}
